// Net_67954972557347
// MI455X (gfx1250) — matrix operand order repaired and checked by reading, not compiled, not run on hardware
//
#include <hip/hip_runtime.h>
#include <hip/hip_bf16.h>

typedef __attribute__((ext_vector_type(16))) _Float16 v16h;
typedef __attribute__((ext_vector_type(8)))  float    v8f;
typedef __attribute__((ext_vector_type(2)))  _Float16 v2h;
typedef __attribute__((ext_vector_type(4)))  _Float16 v4h;

#define LDS_S 164   // LDS row stride in halves (breaks bank-conflict pattern)
#define WAVES 4     // waves per block; each wave owns 16 rows

// v_cvt_pk_rtz_f16_f32 returns an __fp16 vec2; bridge to _Float16 vec2.
__device__ __forceinline__ v2h cvt_pk_h2(float x, float y) {
  auto r = __builtin_amdgcn_cvt_pkrtz(x, y);
  union { decltype(r) a; v2h b; } u;
  u.a = r;
  return u.b;
}

// ---------------------------------------------------------------------------
// Weight packing. Bias is FOLDED into the weights as row k == Kin (the input
// activations carry a constant 1.0 in that column), so no bias array and no
// per-tile bias add in the main kernel.
// Tile t = 32x16 f16 B-fragment, lane-major: lane L's 16 halves contiguous at
// t*512 + L*16.  B layout: lane L -> N = L%16, g = L/16; half j (v=j/2,h=j%2)
// -> K = kk*32 + (j<8 ? 8*g + 2*v + h : 16 + 8*g + (j-8)).
// ---------------------------------------------------------------------------
struct LayerMeta {
  const float* W;
  const float* Bv;
  int Kin, Nout, ksteps, ntiles, tileOfs;
};
struct PackArgs { LayerMeta L[11]; };

#define N_TILES_TOTAL 74

__global__ void pack_weights(PackArgs pa, _Float16* __restrict__ tiles) {
  int gid = blockIdx.x * blockDim.x + threadIdx.x;
  if (gid >= N_TILES_TOTAL * 32) return;
  int t = gid >> 5, lane = gid & 31;
  int l = 0;
  for (int i = 1; i < 11; ++i) if (t >= pa.L[i].tileOfs) l = i;
  const LayerMeta& L = pa.L[l];
  int tt = t - L.tileOfs;
  int kk = tt / L.ntiles;
  int nn = tt - kk * L.ntiles;
  int n  = nn * 16 + (lane & 15);
  int g  = lane >> 4;
  _Float16* dst = tiles + (size_t)t * 512 + (size_t)lane * 16;
  for (int j = 0; j < 16; ++j) {
    int v = j >> 1, h = j & 1;
    int k = kk * 32 + (j < 8 ? g * 8 + 2 * v + h : 16 + g * 8 + (j - 8));
    float val = 0.0f;
    if (n < L.Nout) {
      if (k < L.Kin)       val = L.W[(size_t)k * L.Nout + n];
      else if (k == L.Kin) val = L.Bv[n];           // folded bias row
    }
    dst[j] = (_Float16)val;
  }
}

// ---------------------------------------------------------------------------
// Wave-local LDS sync (per-wave buffers, no __syncthreads needed).
// ---------------------------------------------------------------------------
#define WSYNC() do {                                   \
    __builtin_amdgcn_wave_barrier();                   \
    asm volatile("s_wait_dscnt 0x0" ::: "memory");     \
    __builtin_amdgcn_wave_barrier();                   \
  } while (0)

// ---------------------------------------------------------------------------
// One dense layer on a 16-row tile in LDS.  A-frag: lane L -> M = L%16,
// g = L/16; u32 word idx of row = kk*16 + 4g + (v<4 ? v : v+4).
// Epilogue: packed f16 leaky-ReLU  max(x, 0.01x)  via cvt_pkrtz + pk_mul +
// pk_max (bias already folded into the matmul).
// ---------------------------------------------------------------------------
__device__ __forceinline__ void layer_wmma(
    const _Float16* __restrict__ src, _Float16* __restrict__ dst,
    int ksteps, int ntiles, int tileOfs, int colLimit, int colOff,
    const _Float16* __restrict__ wt, int lane)
{
  const int m = lane & 15, g = lane >> 4;
  const unsigned* sp = (const unsigned*)(src + m * LDS_S);
  for (int nn = 0; nn < ntiles; ++nn) {
    v8f acc = {};
    for (int kk = 0; kk < ksteps; ++kk) {
      union { v16h h; unsigned u[8]; } A;
      int kb = kk * 16 + g * 4;
      A.u[0] = sp[kb + 0];  A.u[1] = sp[kb + 1];
      A.u[2] = sp[kb + 2];  A.u[3] = sp[kb + 3];
      A.u[4] = sp[kb + 8];  A.u[5] = sp[kb + 9];
      A.u[6] = sp[kb + 10]; A.u[7] = sp[kb + 11];
      const v16h* bp = (const v16h*)(wt +
          (size_t)(tileOfs + kk * ntiles + nn) * 512 + (size_t)lane * 16);
      v16h Bf = *bp;
      acc = __builtin_amdgcn_wmma_f32_16x16x32_f16(
          false, A.h, false, Bf, (short)0, acc, false, false);
    }
    int col = nn * 16 + m;
    if (col < colLimit) {
      _Float16* dp = dst + colOff + col;
      const v2h slope = {(_Float16)0.01f, (_Float16)0.01f};
      #pragma unroll
      for (int j = 0; j < 4; ++j) {
        v2h p = cvt_pk_h2(acc[2 * j], acc[2 * j + 1]);
        v2h t = p * slope;                       // v_pk_mul_f16
        v2h r = __builtin_elementwise_max(p, t); // v_pk_max_num_f16
        dp[(2 * j + 8 * g) * LDS_S]     = r[0];
        dp[(2 * j + 1 + 8 * g) * LDS_S] = r[1];
      }
    }
  }
}

// ---------------------------------------------------------------------------
// Main fused MLP kernel: 4 waves/block, each wave runs 16 rows end-to-end in
// private LDS ping-pong buffers.
// ---------------------------------------------------------------------------
__global__ void __launch_bounds__(WAVES * 32)
mlp_main(const float* __restrict__ a, const float* __restrict__ b,
         const float* __restrict__ meta, const float* __restrict__ W9,
         const float* __restrict__ B9, const _Float16* __restrict__ wt,
         float* __restrict__ out)
{
  __shared__ __align__(16) _Float16 smem[WAVES][2][16 * LDS_S];
  const int lane = threadIdx.x & 31;
  const int wid  = threadIdx.x >> 5;
  const int rowBase = blockIdx.x * (WAVES * 16) + wid * 16;
  _Float16* bufA = smem[wid][0];
  _Float16* bufB = smem[wid][1];

  // stage 'a': 16 rows x 16 slots x 4 cols -> bufA cols 0..63
  // (col 44 = data, col 45 = 1.0 bias-column, 46..63 = 0)
  for (int i = lane; i < 16 * 16; i += 32) {
    int r = i >> 4, q = i & 15;
    const float* ap = a + (size_t)(rowBase + r) * 45;
    float f0 = 0.f, f1 = 0.f, f2 = 0.f, f3 = 0.f;
    if (q < 11) {
      int c = q * 4;
      f0 = ap[c]; f1 = ap[c + 1]; f2 = ap[c + 2]; f3 = ap[c + 3];
    } else if (q == 11) {
      f0 = ap[44]; f1 = 1.0f;
    }
    union { v4h h4; v2h h2[2]; } w;
    w.h2[0] = cvt_pk_h2(f0, f1);
    w.h2[1] = cvt_pk_h2(f2, f3);
    *(v4h*)(bufA + r * LDS_S + q * 4) = w.h4;
  }
  WSYNC();
  // a1 = lrelu(a @ Wa + ba) -> bufB cols 0..67   (K=46->64, N=68)
  layer_wmma(bufA, bufB, 2, 5, 0, 68, 0, wt, lane);
  WSYNC();

  // stage 'b': 16 rows x 32 slots x 4 cols -> bufA cols 0..127
  // (cols 100..101 data, col 102 = 1.0, 103..127 = 0)
  for (int i = lane; i < 16 * 32; i += 32) {
    int r = i >> 5, q = i & 31;
    const float* bp = b + (size_t)(rowBase + r) * 102;
    float f0 = 0.f, f1 = 0.f, f2 = 0.f, f3 = 0.f;
    if (q < 25) {
      int c = q * 4;
      f0 = bp[c]; f1 = bp[c + 1]; f2 = bp[c + 2]; f3 = bp[c + 3];
    } else if (q == 25) {
      f0 = bp[100]; f1 = bp[101]; f2 = 1.0f;
    }
    union { v4h h4; v2h h2[2]; } w;
    w.h2[0] = cvt_pk_h2(f0, f1);
    w.h2[1] = cvt_pk_h2(f2, f3);
    *(v4h*)(bufA + r * LDS_S + q * 4) = w.h4;
  }
  WSYNC();
  // b1 = lrelu(b @ Wb + bb) -> bufB cols 68..135 (K=103->128, N=68)
  layer_wmma(bufA, bufB, 4, 5, 10, 68, 68, wt, lane);

  // concat tail: meta -> cols 136..139, ones col 140 = 1.0, 141..159 = 0
  if (lane < 16) {
    const float* mp = meta + (size_t)(rowBase + lane) * 4;
    union { v4h h4; v2h h2[2]; } w;
    w.h2[0] = cvt_pk_h2(mp[0], mp[1]);
    w.h2[1] = cvt_pk_h2(mp[2], mp[3]);
    _Float16* rp = bufB + lane * LDS_S;
    *(v4h*)(rp + 136) = w.h4;
    v4h onesv = {(_Float16)1.0f, (_Float16)0.0f, (_Float16)0.0f, (_Float16)0.0f};
    *(v4h*)(rp + 140) = onesv;
    v4h z = {};
    *(v4h*)(rp + 144) = z;
    *(v4h*)(rp + 148) = z;
    *(v4h*)(rp + 152) = z;
    *(v4h*)(rp + 156) = z;
  }
  WSYNC();

  // tower W0..W8; after each layer write the 1.0 bias column for the next
  const int KS[9]  = {5, 2, 2, 1, 1, 1, 1, 1, 1};
  const int NT[9]  = {4, 4, 2, 2, 2, 2, 2, 2, 2};
  const int TO[9]  = {30, 50, 58, 62, 64, 66, 68, 70, 72};
  const int ONE[9] = {34, 34, 20, 20, 20, 20, 20, 5, -1};
  _Float16* cur = bufB;
  _Float16* nxt = bufA;
  #pragma unroll
  for (int l = 0; l < 9; ++l) {
    layer_wmma(cur, nxt, KS[l], NT[l], TO[l], NT[l] * 16, 0, wt, lane);
    if (ONE[l] >= 0 && lane < 16)
      nxt[lane * LDS_S + ONE[l]] = (_Float16)1.0f;
    WSYNC();
    _Float16* tmp = cur; cur = nxt; nxt = tmp;
  }

  // final 2->1 layer in fp32 (scalar; keeps output full precision)
  if (lane < 16) {
    float x0 = (float)cur[lane * LDS_S + 0];
    float x1 = (float)cur[lane * LDS_S + 1];
    float y  = x0 * W9[0] + x1 * W9[1] + B9[0];
    y = (y > 0.0f) ? y : 0.01f * y;
    out[rowBase + lane] = y;
  }
}

// ---------------------------------------------------------------------------
extern "C" void kernel_launch(void* const* d_in, const int* in_sizes, int n_in,
                              void* d_out, int out_size, void* d_ws, size_t ws_size,
                              hipStream_t stream) {
  const float* a    = (const float*)d_in[0];
  const float* b    = (const float*)d_in[1];
  const float* meta = (const float*)d_in[2];
  const float* Wa   = (const float*)d_in[3];
  const float* ba   = (const float*)d_in[4];
  const float* Wb   = (const float*)d_in[5];
  const float* bb   = (const float*)d_in[6];
  const float* W[10];
  const float* Bv[10];
  for (int i = 0; i < 10; ++i) {
    W[i]  = (const float*)d_in[7 + 2 * i];
    Bv[i] = (const float*)d_in[8 + 2 * i];
  }

  _Float16* tiles = (_Float16*)d_ws;

  // layer table: Wa, Wb, W0..W8 (W9 handled scalar in the main kernel)
  const float* Wp[11] = {Wa, Wb, W[0], W[1], W[2], W[3], W[4], W[5], W[6], W[7], W[8]};
  const float* Bp[11] = {ba, bb, Bv[0], Bv[1], Bv[2], Bv[3], Bv[4], Bv[5], Bv[6], Bv[7], Bv[8]};
  const int Kin[11]  = {45, 102, 140, 34, 34, 20, 20, 20, 20, 20, 5};
  const int Nout[11] = {68, 68, 34, 34, 20, 20, 20, 20, 20, 5, 2};
  const int ks[11]   = {2, 4, 5, 2, 2, 1, 1, 1, 1, 1, 1};
  const int nt[11]   = {5, 5, 4, 4, 2, 2, 2, 2, 2, 2, 2};
  const int to[11]   = {0, 10, 30, 50, 58, 62, 64, 66, 68, 70, 72};

  PackArgs pa;
  for (int i = 0; i < 11; ++i) {
    pa.L[i].W = Wp[i];   pa.L[i].Bv = Bp[i];
    pa.L[i].Kin = Kin[i]; pa.L[i].Nout = Nout[i];
    pa.L[i].ksteps = ks[i]; pa.L[i].ntiles = nt[i];
    pa.L[i].tileOfs = to[i];
  }

  {
    int threads = N_TILES_TOTAL * 32;
    int blk = 128;
    pack_weights<<<(threads + blk - 1) / blk, blk, 0, stream>>>(pa, tiles);
  }

  int nB = in_sizes[0] / 45;              // 262144
  int rowsPerBlock = WAVES * 16;          // 64
  int grid = nB / rowsPerBlock;           // 4096
  mlp_main<<<grid, WAVES * 32, 0, stream>>>(a, b, meta, W[9], Bv[9],
                                            tiles, (float*)d_out);
}
